// LSTM_SOFT_ATT_STACK_57990648430685
// MI455X (gfx1250) — matrix operand order repaired and checked by reading, not compiled, not run on hardware
//
#include <hip/hip_runtime.h>
#include <hip/hip_bf16.h>
#include <math.h>

#define BDIM 256
#define RDIM 512
#define ADIM 196
#define VDIM 10000
#define LSTEPS 2

typedef __attribute__((ext_vector_type(16))) __bf16 v16bf;
typedef __attribute__((ext_vector_type(4)))  __bf16 v4bf;
typedef __attribute__((ext_vector_type(8)))  float  v8f;

__device__ __forceinline__ float sigmoidf_(float x) { return 1.0f / (1.0f + __expf(-x)); }

__device__ __forceinline__ v16bf cvt16_bf16(float4 a, float4 b, float4 c, float4 d) {
  v16bf r;
  r[0]  = (__bf16)a.x; r[1]  = (__bf16)a.y; r[2]  = (__bf16)a.z; r[3]  = (__bf16)a.w;
  r[4]  = (__bf16)b.x; r[5]  = (__bf16)b.y; r[6]  = (__bf16)b.z; r[7]  = (__bf16)b.w;
  r[8]  = (__bf16)c.x; r[9]  = (__bf16)c.y; r[10] = (__bf16)c.z; r[11] = (__bf16)c.w;
  r[12] = (__bf16)d.x; r[13] = (__bf16)d.y; r[14] = (__bf16)d.z; r[15] = (__bf16)d.w;
  return r;
}

// One-time f32 -> bf16 weight conversion.
__global__ void cvt_f32_bf16_kernel(const float* __restrict__ src,
                                    __bf16* __restrict__ dst, int n) {
  int i = (blockIdx.x * blockDim.x + threadIdx.x) * 4;
  if (i < n) {
    float4 v = *(const float4*)(src + i);
    v4bf o;
    o[0] = (__bf16)v.x; o[1] = (__bf16)v.y; o[2] = (__bf16)v.z; o[3] = (__bf16)v.w;
    *(v4bf*)(dst + i) = o;
  }
}

// C(MxN) = A(MxKK f32 row-major) @ Wb(NxKK bf16 row-major)^T [+ bias] [+ C_old].
// One wave computes 16(M) x 16*NT(N) with v_wmma_f32_16x16x32_bf16.
//  - B tiles use 32-bit byte offsets against a uniform base (saddr+voffset form,
//    no 64-bit per-load address math).
//  - Two named B buffers processed in interleaved pairs force one B-fragment
//    load to stay in flight across every WMMA (real load/MMA overlap).
template<int NT, int KK>
__launch_bounds__(128)
__global__ void gemm_bf16w_wmma(const float* __restrict__ A, int lda,
                                const __bf16* __restrict__ Wb,   // (N x KK) bf16
                                const float* __restrict__ bias,  // (N) or null
                                float* __restrict__ C, int ldc,
                                int M, int N, int accumulate)
{
  const int lane = threadIdx.x & 31;
  const int wave = threadIdx.x >> 5;
  const int lr   = lane & 15;
  const int half = lane >> 4;
  const int mBase  = (blockIdx.x * 4 + wave) * 16;
  const int nBase0 = blockIdx.y * (16 * NT);
  if (mBase >= M) return;  // wave-uniform

  const float* Arow = A + (size_t)(mBase + lr) * lda;

  // 32-bit byte offsets of each B fragment's base within Wb (clamped for edge
  // tiles; the corresponding stores are masked). Fits: N*KK*2 <= ~10.2 MB.
  unsigned int bOff[NT];
  #pragma unroll
  for (int t = 0; t < NT; ++t) {
    int n = nBase0 + t * 16 + lr;
    if (n >= N) n = N - 1;
    bOff[t] = (unsigned int)((n * KK + half * 16) * 2);
  }

  v8f acc[NT];
  #pragma unroll
  for (int t = 0; t < NT; ++t) {
    int n  = nBase0 + t * 16 + lr;
    int nc = (n < N) ? n : (N - 1);
    float bv = bias ? bias[nc] : 0.0f;
    #pragma unroll
    for (int v = 0; v < 8; ++v) {
      float init = bv;
      if (accumulate) init += C[(size_t)(mBase + v + half * 8) * ldc + nc];
      acc[t][v] = init;
    }
  }

  const char* wb = (const char*)Wb;

  for (int kb = 0; kb < KK; kb += 32) {
    // A fragment (16-bit A 16x32 layout matching B: half 0 lanes hold K 0..15,
    // half 1 lanes hold K 16..31): f32 loads + register convert.
    const float* ap = Arow + kb + half * 16;
    float4 a0 = ((const float4*)ap)[0];
    float4 a1 = ((const float4*)ap)[1];
    float4 a2 = ((const float4*)ap)[2];
    float4 a3 = ((const float4*)ap)[3];
    v16bf af = cvt16_bf16(a0, a1, a2, a3);

    // B fragments (32x16 bf16: lane = column n; half 0 = K 0..15, half 1 =
    // K 16..31): single 32B vector load each, two-deep pipeline.
    const char* wkb = wb + kb * 2;
    v16bf b0 = *(const v16bf*)(wkb + bOff[0]);
    v16bf b1 = b0;
    if (NT > 1) b1 = *(const v16bf*)(wkb + bOff[1]);

    #pragma unroll
    for (int t = 0; t < NT; t += 2) {
      acc[t] = __builtin_amdgcn_wmma_f32_16x16x32_bf16(
          false, af, false, b0, (short)0, acc[t], false, false);
      if (t + 2 < NT) b0 = *(const v16bf*)(wkb + bOff[t + 2]);
      if (t + 1 < NT) {
        acc[t + 1] = __builtin_amdgcn_wmma_f32_16x16x32_bf16(
            false, af, false, b1, (short)0, acc[t + 1], false, false);
        if (t + 3 < NT) b1 = *(const v16bf*)(wkb + bOff[t + 3]);
      }
    }
  }

  // C/D layout: VGPR v, lane l -> row mBase + v + 8*half, col nBase + 16*t + lr
  #pragma unroll
  for (int t = 0; t < NT; ++t) {
    int n = nBase0 + t * 16 + lr;
    if (n < N) {
      #pragma unroll
      for (int v = 0; v < 8; ++v)
        C[(size_t)(mBase + v + half * 8) * ldc + n] = acc[t][v];
    }
  }
}

// score[b,i] = sum_a tanh(att_v[b,i,a] + att_h[b,i]) * wd2d[a] + bd2d
__global__ void score_kernel(const float* __restrict__ att_v,
                             const float* __restrict__ att_h,
                             const float* __restrict__ wd2d,
                             const float* __restrict__ bd2d,
                             float* __restrict__ score)
{
  int idx = blockIdx.x * blockDim.x + threadIdx.x;   // b*ADIM + i
  if (idx >= BDIM * ADIM) return;
  float h = att_h[idx];
  const float* row = att_v + (size_t)idx * ADIM;
  float s = 0.0f;
  for (int a = 0; a < ADIM; ++a) s += tanhf(row[a] + h) * wd2d[a];
  score[idx] = s + bd2d[0];
}

// Per batch row: w = softmax(score[b,:]); att_res[b,r] = sum_i att[b,i,r]*w[i]
__global__ void softmax_attres_kernel(const float* __restrict__ score,
                                      const float* __restrict__ att,
                                      float* __restrict__ att_res)
{
  __shared__ float w[ADIM];
  __shared__ float red[256];
  int b = blockIdx.x, t = threadIdx.x;

  float v = (t < ADIM) ? score[b * ADIM + t] : -INFINITY;
  red[t] = v; __syncthreads();
  for (int s = 128; s > 0; s >>= 1) { if (t < s) red[t] = fmaxf(red[t], red[t + s]); __syncthreads(); }
  float mx = red[0]; __syncthreads();

  float e = (t < ADIM) ? __expf(v - mx) : 0.0f;
  red[t] = e; __syncthreads();
  for (int s = 128; s > 0; s >>= 1) { if (t < s) red[t] += red[t + s]; __syncthreads(); }
  float inv = 1.0f / red[0];
  if (t < ADIM) w[t] = e * inv;
  __syncthreads();

  const float* ab = att + (size_t)b * ADIM * RDIM;
  for (int r = t; r < RDIM; r += 256) {
    float acc = 0.0f;
    for (int i = 0; i < ADIM; ++i) acc += ab[(size_t)i * RDIM + r] * w[i];
    att_res[b * RDIM + r] = acc;
  }
}

// LSTM gate nonlinearities + state update
__global__ void lstm_pointwise_kernel(const float* __restrict__ sums,
                                      const float* __restrict__ prev_c,
                                      float* __restrict__ next_c,
                                      float* __restrict__ next_h)
{
  int idx = blockIdx.x * blockDim.x + threadIdx.x;   // b*RDIM + r
  if (idx >= BDIM * RDIM) return;
  int b = idx / RDIM, r = idx % RDIM;
  const float* srow = sums + (size_t)b * (4 * RDIM);
  float ig = sigmoidf_(srow[r]);
  float fg = sigmoidf_(srow[RDIM + r]);
  float og = sigmoidf_(srow[2 * RDIM + r]);
  float it = tanhf(srow[3 * RDIM + r]);
  float c  = fg * prev_c[idx] + ig * it;
  float h  = og * tanhf(c);
  next_c[idx] = c;
  next_h[idx] = h;
}

__global__ void logsoftmax_kernel(const float* __restrict__ logits,
                                  float* __restrict__ out)
{
  __shared__ float red[256];
  int b = blockIdx.x, t = threadIdx.x;
  const float* row = logits + (size_t)b * VDIM;

  float mx = -INFINITY;
  for (int v = t; v < VDIM; v += 256) mx = fmaxf(mx, row[v]);
  red[t] = mx; __syncthreads();
  for (int s = 128; s > 0; s >>= 1) { if (t < s) red[t] = fmaxf(red[t], red[t + s]); __syncthreads(); }
  mx = red[0]; __syncthreads();

  float sum = 0.0f;
  for (int v = t; v < VDIM; v += 256) sum += __expf(row[v] - mx);
  red[t] = sum; __syncthreads();
  for (int s = 128; s > 0; s >>= 1) { if (t < s) red[t] += red[t + s]; __syncthreads(); }
  float lse = mx + __logf(red[0]);

  for (int v = t; v < VDIM; v += 256) out[(size_t)b * VDIM + v] = row[v] - lse;
}

static inline int cdiv(int a, int b) { return (a + b - 1) / b; }

extern "C" void kernel_launch(void* const* d_in, const int* in_sizes, int n_in,
                              void* d_out, int out_size, void* d_ws, size_t ws_size,
                              hipStream_t stream)
{
  const float* x      = (const float*)d_in[0];
  const float* att    = (const float*)d_in[1];
  const float* inputs = (const float*)d_in[2];
  const float* Wa2a   = (const float*)d_in[3];
  const float* ba2a   = (const float*)d_in[4];
  const float* Wh2a   = (const float*)d_in[5];
  const float* bh2a   = (const float*)d_in[6];
  const float* wd2d   = (const float*)d_in[7];
  const float* bd2d   = (const float*)d_in[8];
  const float* Wi2h   = (const float*)d_in[9];
  const float* bi2h   = (const float*)d_in[10];
  const float* Wh2h   = (const float*)d_in[11];
  const float* bh2h   = (const float*)d_in[12];
  const float* Wr2a   = (const float*)d_in[13];
  const float* br2a   = (const float*)d_in[14];
  const float* Wproj  = (const float*)d_in[15];
  const float* bproj  = (const float*)d_in[16];

  // --- Workspace carve-up ---
  float* ws      = (float*)d_ws;
  float* att_v   = ws;                                          // B*A*A
  float* att_h   = att_v + (size_t)BDIM * ADIM * ADIM;          // B*A
  float* score   = att_h + (size_t)BDIM * ADIM;                 // B*A
  float* att_res = score + (size_t)BDIM * ADIM;                 // B*R
  float* sums    = att_res + (size_t)BDIM * RDIM;               // B*4R
  float* logits  = sums + (size_t)BDIM * 4 * RDIM;              // B*V
  __bf16* bfbase   = (__bf16*)(logits + (size_t)BDIM * VDIM);
  __bf16* Wa2a_bf  = bfbase;                                     // A*R
  __bf16* Wh2a_bf  = Wa2a_bf + (size_t)ADIM * RDIM;              // A*R
  __bf16* Wi2h_bf  = Wh2a_bf + (size_t)ADIM * RDIM;              // 4R*R
  __bf16* Wh2h_bf  = Wi2h_bf + (size_t)4 * RDIM * RDIM;          // 4R*R
  __bf16* Wr2a_bf  = Wh2h_bf + (size_t)4 * RDIM * RDIM;          // 4R*R
  __bf16* Wproj_bf = Wr2a_bf + (size_t)4 * RDIM * RDIM;          // V*R

  // --- One-time weight conversions ---
  {
    struct { const float* s; __bf16* d; int n; } cv[6] = {
      { Wa2a,  Wa2a_bf,  ADIM * RDIM },
      { Wh2a,  Wh2a_bf,  ADIM * RDIM },
      { Wi2h,  Wi2h_bf,  4 * RDIM * RDIM },
      { Wh2h,  Wh2h_bf,  4 * RDIM * RDIM },
      { Wr2a,  Wr2a_bf,  4 * RDIM * RDIM },
      { Wproj, Wproj_bf, VDIM * RDIM },
    };
    for (int i = 0; i < 6; ++i)
      cvt_f32_bf16_kernel<<<cdiv(cv[i].n / 4, 256), 256, 0, stream>>>(cv[i].s, cv[i].d, cv[i].n);
  }

  // Output layout: states [c0,h0,c1,h1] (4*B*R) then logsoft (B*V)
  float* out = (float*)d_out;
  float* st_c[LSTEPS] = { out,                 out + 2 * BDIM * RDIM };
  float* st_h[LSTEPS] = { out + BDIM * RDIM,   out + 3 * BDIM * RDIM };
  float* logsoft = out + 4 * BDIM * RDIM;

  // att_v is loop-invariant: compute ONCE. M=B*A=50176, N=196, K=512
  {
    int M = BDIM * ADIM;
    gemm_bf16w_wmma<13, RDIM><<<dim3(M / 64, 1), 128, 0, stream>>>(
        att, RDIM, Wa2a_bf, ba2a, att_v, ADIM, M, ADIM, 0);
  }

  const float* cur = x;
  for (int i = 0; i < LSTEPS; ++i) {
    const float* prev_c = inputs + (size_t)(2 * i) * BDIM * RDIM;
    const float* prev_h = inputs + (size_t)(2 * i + 1) * BDIM * RDIM;

    // att_h = prev_h @ Wh2a^T + bh2a    (256 x 196)
    gemm_bf16w_wmma<13, RDIM><<<dim3(BDIM / 64, 1), 128, 0, stream>>>(
        prev_h, RDIM, Wh2a_bf, bh2a, att_h, ADIM, BDIM, ADIM, 0);

    // score = tanh(att_v + att_h) . wd2d + bd2d
    score_kernel<<<(BDIM * ADIM) / 256, 256, 0, stream>>>(att_v, att_h, wd2d, bd2d, score);

    // softmax over positions + weighted sum of att rows (att hits L2)
    softmax_attres_kernel<<<BDIM, 256, 0, stream>>>(score, att, att_res);

    // sums = cur@Wi2h^T + bi2h + prev_h@Wh2h^T + bh2h + att_res@Wr2a^T + br2a
    gemm_bf16w_wmma<4, RDIM><<<dim3(BDIM / 64, (4 * RDIM) / 64), 128, 0, stream>>>(
        cur, RDIM, Wi2h_bf, bi2h, sums, 4 * RDIM, BDIM, 4 * RDIM, 0);
    gemm_bf16w_wmma<4, RDIM><<<dim3(BDIM / 64, (4 * RDIM) / 64), 128, 0, stream>>>(
        prev_h, RDIM, Wh2h_bf, bh2h, sums, 4 * RDIM, BDIM, 4 * RDIM, 1);
    gemm_bf16w_wmma<4, RDIM><<<dim3(BDIM / 64, (4 * RDIM) / 64), 128, 0, stream>>>(
        att_res, RDIM, Wr2a_bf, br2a, sums, 4 * RDIM, BDIM, 4 * RDIM, 1);

    // gates + state update -> writes states directly into d_out
    lstm_pointwise_kernel<<<(BDIM * RDIM) / 256, 256, 0, stream>>>(
        sums, prev_c, st_c[i], st_h[i]);

    cur = st_h[i];
  }

  // logits = h_last @ Wproj^T + bproj   (256 x 10000, NT=5 -> 125 col groups)
  gemm_bf16w_wmma<5, RDIM><<<dim3(BDIM / 64, VDIM / 80), 128, 0, stream>>>(
      cur, RDIM, Wproj_bf, bproj, logits, VDIM, BDIM, VDIM, 0);

  logsoftmax_kernel<<<BDIM, 256, 0, stream>>>(logits, logsoft);
}
